// SUNConv_23184233463964
// MI455X (gfx1250) — matrix operand order repaired and checked by reading, not compiled, not run on hardware
//
#include <hip/hip_runtime.h>

// ---------------------------------------------------------------------------
// SUNConv for MI455X (gfx1250, wave32, WMMA bf16 16x16x32, f32 accumulate)
// fp32 handled via 2-term bf16 split (hi+lo), 3 WMMAs per tile product
// ---------------------------------------------------------------------------

#define LDP 129   // padded LDS row stride (floats) for 128x128 f32 tiles
#define LDK 136   // padded LDS row stride (bf16) for transposed Xl tile

typedef __attribute__((ext_vector_type(16))) __bf16 bf16x16;
typedef __attribute__((ext_vector_type(8)))  __bf16 bf16x8;
typedef __attribute__((ext_vector_type(8)))  float  floatx8;

__device__ __forceinline__ floatx8 mma3(floatx8 c, bf16x16 ah, bf16x16 al,
                                        bf16x16 bh, bf16x16 bl) {
  // (Ah+Al)*(Bh+Bl) ~= Ah*Bh + Ah*Bl + Al*Bh   (drop Al*Bl, ~2^-16 rel)
  c = __builtin_amdgcn_wmma_f32_16x16x32_bf16(false, ah, false, bh, (short)0, c, false, false);
  c = __builtin_amdgcn_wmma_f32_16x16x32_bf16(false, ah, false, bl, (short)0, c, false, false);
  c = __builtin_amdgcn_wmma_f32_16x16x32_bf16(false, al, false, bh, (short)0, c, false, false);
  return c;
}

// A fragment (16x32 bf16): rowPtr = src + (Mbase + r)*ld  (per-lane row base).
// elements 0..7  <-> K = k0 + 16*half + e
// elements 8..15 <-> K = k0 + 16*half + 8 + e
__device__ __forceinline__ void load_afrag_f32(const float* rowPtr, int k0, int half,
                                               bf16x16& ah, bf16x16& al) {
  const float* p0 = rowPtr + k0 + 16 * half;
  const float* p1 = p0 + 8;
#pragma unroll
  for (int e = 0; e < 8; ++e) {
    float x0 = p0[e];
    __bf16 h0 = (__bf16)x0;
    ah[e] = h0;
    al[e] = (__bf16)(x0 - (float)h0);
    float x1 = p1[e];
    __bf16 h1 = (__bf16)x1;
    ah[8 + e] = h1;
    al[8 + e] = (__bf16)(x1 - (float)h1);
  }
}

// B fragment (32x16 bf16) from pre-transposed bf16 weights wt[n][k] (ldk elems):
// per lane: N = n, elements e=0..15 <-> K = k0 + 16*half + e (contiguous 32B)
__device__ __forceinline__ bf16x16 load_bfrag_bf16(const __bf16* wt, int ldk,
                                                   int n, int k0, int half) {
  return *(const bf16x16*)(wt + (size_t)n * ldk + k0 + 16 * half);
}

// B fragment from pre-split transposed bf16 tile in LDS (16B-aligned rows):
// two 16-byte halves joined (row stride LDK=136 elems => rows are 16B aligned)
__device__ __forceinline__ bf16x16 load_bfrag_lds_bf16(const __bf16* base) {
  bf16x8 lo = *(const bf16x8*)(base);
  bf16x8 hi = *(const bf16x8*)(base + 8);
  return __builtin_shufflevector(lo, hi, 0, 1, 2, 3, 4, 5, 6, 7,
                                 8, 9, 10, 11, 12, 13, 14, 15);
}

// ---------------------------------------------------------------------------
// Weight prep: transpose + hi/lo bf16 split.  W (KxN row-major f32) -> wt[n][k]
// ---------------------------------------------------------------------------
__global__ void prep_weights(const float* __restrict__ W0, const float* __restrict__ W10,
                             const float* __restrict__ W2,
                             __bf16* w0h, __bf16* w0l, __bf16* w10h, __bf16* w10l,
                             __bf16* w2h, __bf16* w2l) {
  int idx = blockIdx.x * 256 + threadIdx.x;
  if (idx < 896 * 128) {                 // W10: K=896, N=128
    int k = idx / 128, n = idx % 128;
    float x = W10[idx];
    __bf16 h = (__bf16)x;
    w10h[n * 896 + k] = h;
    w10l[n * 896 + k] = (__bf16)(x - (float)h);
  }
  if (idx < 128 * 128) {                 // W0 and W2: K=128, N=128
    int k = idx / 128, n = idx % 128;
    float x0 = W0[idx];
    __bf16 h0 = (__bf16)x0;
    w0h[n * 128 + k] = h0;
    w0l[n * 128 + k] = (__bf16)(x0 - (float)h0);
    float x2 = W2[idx];
    __bf16 h2 = (__bf16)x2;
    w2h[n * 128 + k] = h2;
    w2l[n * 128 + k] = (__bf16)(x2 - (float)h2);
  }
}

// dst[b,j,d] = mean_i src[b,i,j,d]   grid=(128,8) block=128
__global__ void reduce_over_i(const float* __restrict__ src, float* __restrict__ dst) {
  int j = blockIdx.x, b = blockIdx.y, d = threadIdx.x;
  const float* p = src + ((size_t)b << 21) + ((size_t)j << 7) + d;
  float s = 0.f;
  for (int i = 0; i < 128; ++i) s += p[(size_t)i << 14];
  dst[((size_t)(b * 128 + j) << 7) + d] = s * (1.f / 128.f);
}

// X6[b,i,d] = mean_j X[b,i,j,d]; Xdiag[b,i,d] = X[b,i,i,d]  grid=(128,8) block=128
__global__ void reduce_over_j_and_diag(const float* __restrict__ X,
                                       float* __restrict__ X6m, float* __restrict__ Xdiag) {
  int i = blockIdx.x, b = blockIdx.y, d = threadIdx.x;
  const float* p = X + ((size_t)(b * 128 + i) << 14) + d;
  float s = 0.f;
  for (int j = 0; j < 128; ++j) s += p[j * 128];
  X6m[((size_t)(b * 128 + i) << 7) + d] = s * (1.f / 128.f);
  Xdiag[((size_t)(b * 128 + i) << 7) + d] = p[i * 128];
}

// ---------------------------------------------------------------------------
// Stage AB: per (b,i):  Xl = relu(X[b,i] @ W0 + b0);  X4[b,i] = A[b] @ Xl
// 256 threads = 8 waves; wave w owns output rows [16w,16w+16), all 128 cols.
// Xl is split to bf16 hi/lo ONCE at store time, stored transposed [d][k] in
// LDS so GEMM2 B-fragments are contiguous 16B ds_load_b128 pairs.
// ---------------------------------------------------------------------------
__global__ void __launch_bounds__(256)
stage_ab(const float* __restrict__ X, const float* __restrict__ A,
         const __bf16* __restrict__ w0h, const __bf16* __restrict__ w0l,
         const float* __restrict__ b0, float* __restrict__ X4ws) {
  extern __shared__ __bf16 smem_ab[];
  __bf16* ldsTh = smem_ab;              // [128][LDK] hi plane, transposed [d][k]
  __bf16* ldsTl = smem_ab + 128 * LDK;  // [128][LDK] lo plane
  const int i = blockIdx.x, b = blockIdx.y;
  const int tid = threadIdx.x;
  const int wave = tid >> 5, lane = tid & 31, r = lane & 15, half = lane >> 4;
  const int Mbase = wave * 16;
  const float* Xt = X + ((size_t)(b * 128 + i) << 14);

  floatx8 acc[8];
#pragma unroll
  for (int nt = 0; nt < 8; ++nt) {
    float bv = b0[nt * 16 + r];
#pragma unroll
    for (int v = 0; v < 8; ++v) acc[nt][v] = bv;
  }

  // ---- GEMM1: Xl = relu(X @ W0 + b0) ----
  const float* arow = Xt + (Mbase + r) * 128;
  for (int kt = 0; kt < 4; ++kt) {
    bf16x16 ah, al;
    load_afrag_f32(arow, kt * 32, half, ah, al);
#pragma unroll
    for (int nt = 0; nt < 8; ++nt) {
      bf16x16 bh = load_bfrag_bf16(w0h, 128, nt * 16 + r, kt * 32, half);
      bf16x16 bl = load_bfrag_bf16(w0l, 128, nt * 16 + r, kt * 32, half);
      acc[nt] = mma3(acc[nt], ah, al, bh, bl);
    }
  }
  // relu + split once + store transposed: element (M=k', N=d') -> ldsT[d'][k']
#pragma unroll
  for (int nt = 0; nt < 8; ++nt)
#pragma unroll
    for (int v = 0; v < 8; ++v) {
      float x = acc[nt][v];
      x = x > 0.f ? x : 0.f;
      __bf16 h = (__bf16)x;
      __bf16 l = (__bf16)(x - (float)h);
      int idx = (nt * 16 + r) * LDK + Mbase + 8 * half + v;
      ldsTh[idx] = h;
      ldsTl[idx] = l;
    }
  __syncthreads();

  // ---- GEMM2: X4 = A[b] @ Xl ----
#pragma unroll
  for (int nt = 0; nt < 8; ++nt)
#pragma unroll
    for (int v = 0; v < 8; ++v) acc[nt][v] = 0.f;
  const float* arow2 = A + ((size_t)b << 14) + (Mbase + r) * 128;
  for (int kt = 0; kt < 4; ++kt) {
    bf16x16 ah, al;
    load_afrag_f32(arow2, kt * 32, half, ah, al);
#pragma unroll
    for (int nt = 0; nt < 8; ++nt) {
      int boff = (nt * 16 + r) * LDK + kt * 32 + 16 * half;
      bf16x16 bh = load_bfrag_lds_bf16(ldsTh + boff);
      bf16x16 bl = load_bfrag_lds_bf16(ldsTl + boff);
      acc[nt] = mma3(acc[nt], ah, al, bh, bl);
    }
  }
  float* X4t = X4ws + ((size_t)(b * 128 + i) << 14);
#pragma unroll
  for (int nt = 0; nt < 8; ++nt)
#pragma unroll
    for (int v = 0; v < 8; ++v)
      X4t[(Mbase + 8 * half + v) * 128 + nt * 16 + r] = acc[nt][v];
}

// ---------------------------------------------------------------------------
// Stage CD: per (b,i):
//   Xm = Xc @ W10 + b10  (Xc assembled virtually from 7 K-blocks, K=896)
//   row i of Xm recomputed with W11/b11 (diagonal)
//   out = relu(Xm @ W2 + b2)
// ---------------------------------------------------------------------------
__global__ void __launch_bounds__(256)
stage_cd(const float* __restrict__ X, const float* __restrict__ X4ws,
         const float* __restrict__ Xdiag, const float* __restrict__ X5m,
         const float* __restrict__ X6m, const float* __restrict__ X7m,
         const __bf16* __restrict__ w10h, const __bf16* __restrict__ w10l,
         const float* __restrict__ b10,
         const float* __restrict__ W11, const float* __restrict__ b11,
         const __bf16* __restrict__ w2h, const __bf16* __restrict__ w2l,
         const float* __restrict__ b2, float* __restrict__ out) {
  extern __shared__ float lds[];
  float* ldsXm = lds;                 // 128*LDP
  float* xcd   = lds + 128 * LDP;     // 896 (diagonal Xc row)
  float* red   = xcd + 896;           // 256 (reduction scratch)

  const int i = blockIdx.x, b = blockIdx.y;
  const int bi = b * 128 + i;
  const int tid = threadIdx.x;
  const int wave = tid >> 5, lane = tid & 31, r = lane & 15, half = lane >> 4;
  const int Mbase = wave * 16;

  floatx8 acc[8];
#pragma unroll
  for (int nt = 0; nt < 8; ++nt) {
    float bv = b10[nt * 16 + r];
#pragma unroll
    for (int v = 0; v < 8; ++v) acc[nt][v] = bv;
  }

  // virtual-concat sources: {ptr, row stride in floats}; stride 0 = j-broadcast
  const float* s0 = X     + ((size_t)bi << 14);  // X[b,i,j,:]
  const float* s1 = Xdiag + ((size_t)bi << 7);   // X2: Xdiag[b,i,:]   (const over j)
  const float* s2 = Xdiag + ((size_t)b << 14);   // X3: Xdiag[b,j,:]
  const float* s3 = X4ws  + ((size_t)bi << 14);  // X4[b,i,j,:]
  const float* s4 = X5m   + ((size_t)b << 14);   // X5: mean_i X [b,j,:]
  const float* s5 = X6m   + ((size_t)bi << 7);   // X6: mean_j X [b,i,:] (const over j)
  const float* s6 = X7m   + ((size_t)b << 14);   // X7: mean_i X4 [b,j,:]

  auto gemm_blk = [&](const float* src, int ld, int blk) {
    const float* rowp = src + (Mbase + r) * ld;
    for (int kt = 0; kt < 4; ++kt) {
      bf16x16 ah, al;
      load_afrag_f32(rowp, kt * 32, half, ah, al);
      int kg = blk * 128 + kt * 32;
#pragma unroll
      for (int nt = 0; nt < 8; ++nt) {
        bf16x16 bh = load_bfrag_bf16(w10h, 896, nt * 16 + r, kg, half);
        bf16x16 bl = load_bfrag_bf16(w10l, 896, nt * 16 + r, kg, half);
        acc[nt] = mma3(acc[nt], ah, al, bh, bl);
      }
    }
  };
  gemm_blk(s0, 128, 0);
  gemm_blk(s1, 0,   1);
  gemm_blk(s2, 128, 2);
  gemm_blk(s3, 128, 3);
  gemm_blk(s4, 128, 4);
  gemm_blk(s5, 0,   5);
  gemm_blk(s6, 128, 6);

#pragma unroll
  for (int nt = 0; nt < 8; ++nt)
#pragma unroll
    for (int v = 0; v < 8; ++v)
      ldsXm[(Mbase + 8 * half + v) * LDP + nt * 16 + r] = acc[nt][v];
  __syncthreads();

  // ---- diagonal row (j == i): Xm[i,:] = Xc_diag @ W11 + b11 (exact f32 VALU) ----
  if (tid < 128) {
    int d = tid;
    float xd = Xdiag[((size_t)bi << 7) + d];
    xcd[d] = xd;          // X  block at j=i is the diagonal
    xcd[128 + d] = xd;    // X2
    xcd[256 + d] = xd;    // X3 at j=i
    xcd[384 + d] = X4ws[((size_t)bi << 14) + i * 128 + d];
    xcd[512 + d] = X5m[((size_t)bi << 7) + d];
    xcd[640 + d] = X6m[((size_t)bi << 7) + d];
    xcd[768 + d] = X7m[((size_t)bi << 7) + d];
  }
  __syncthreads();
  {
    int n = tid & 127, hp = tid >> 7;
    float s = 0.f;
    for (int k = hp * 448; k < hp * 448 + 448; ++k) s += xcd[k] * W11[k * 128 + n];
    red[tid] = s;
  }
  __syncthreads();
  if (tid < 128) ldsXm[i * LDP + tid] = red[tid] + red[tid + 128] + b11[tid];
  __syncthreads();

  // ---- GEMM4: out = relu(Xm @ W2 + b2) ----
#pragma unroll
  for (int nt = 0; nt < 8; ++nt) {
    float bv = b2[nt * 16 + r];
#pragma unroll
    for (int v = 0; v < 8; ++v) acc[nt][v] = bv;
  }
  const float* mrow = ldsXm + (Mbase + r) * LDP;
  for (int kt = 0; kt < 4; ++kt) {
    bf16x16 ah, al;
    load_afrag_f32(mrow, kt * 32, half, ah, al);
#pragma unroll
    for (int nt = 0; nt < 8; ++nt) {
      bf16x16 bh = load_bfrag_bf16(w2h, 128, nt * 16 + r, kt * 32, half);
      bf16x16 bl = load_bfrag_bf16(w2l, 128, nt * 16 + r, kt * 32, half);
      acc[nt] = mma3(acc[nt], ah, al, bh, bl);
    }
  }
  float* ot = out + ((size_t)bi << 14);
#pragma unroll
  for (int nt = 0; nt < 8; ++nt)
#pragma unroll
    for (int v = 0; v < 8; ++v) {
      float x = acc[nt][v];
      ot[(Mbase + 8 * half + v) * 128 + nt * 16 + r] = x > 0.f ? x : 0.f;
    }
}

// ---------------------------------------------------------------------------
extern "C" void kernel_launch(void* const* d_in, const int* in_sizes, int n_in,
                              void* d_out, int out_size, void* d_ws, size_t ws_size,
                              hipStream_t stream) {
  const float* A   = (const float*)d_in[0];
  const float* X   = (const float*)d_in[1];
  const float* W0  = (const float*)d_in[2];
  const float* b0  = (const float*)d_in[3];
  const float* W10 = (const float*)d_in[4];
  const float* b10 = (const float*)d_in[5];
  const float* W11 = (const float*)d_in[6];
  const float* b11 = (const float*)d_in[7];
  const float* W2  = (const float*)d_in[8];
  const float* b2  = (const float*)d_in[9];
  float* out = (float*)d_out;

  char* ws = (char*)d_ws;
  size_t off = 0;
  auto alloc = [&](size_t bytes) -> void* {
    void* p = ws + off;
    off = (off + bytes + 255) & ~(size_t)255;
    return p;
  };
  float* X4ws  = (float*)alloc((size_t)8 * 128 * 128 * 128 * 4);  // 67 MB
  float* X5m   = (float*)alloc((size_t)8 * 128 * 128 * 4);
  float* X6m   = (float*)alloc((size_t)8 * 128 * 128 * 4);
  float* X7m   = (float*)alloc((size_t)8 * 128 * 128 * 4);
  float* Xdiag = (float*)alloc((size_t)8 * 128 * 128 * 4);
  __bf16* w0h  = (__bf16*)alloc(128 * 128 * 2);
  __bf16* w0l  = (__bf16*)alloc(128 * 128 * 2);
  __bf16* w10h = (__bf16*)alloc(896 * 128 * 2);
  __bf16* w10l = (__bf16*)alloc(896 * 128 * 2);
  __bf16* w2h  = (__bf16*)alloc(128 * 128 * 2);
  __bf16* w2l  = (__bf16*)alloc(128 * 128 * 2);

  prep_weights<<<dim3(448), 256, 0, stream>>>(W0, W10, W2, w0h, w0l, w10h, w10l, w2h, w2l);
  reduce_over_i<<<dim3(128, 8), 128, 0, stream>>>(X, X5m);
  reduce_over_j_and_diag<<<dim3(128, 8), 128, 0, stream>>>(X, X6m, Xdiag);

  size_t lds_ab = (size_t)128 * LDK * 2 * sizeof(__bf16);  // hi + lo planes
  stage_ab<<<dim3(128, 8), 256, lds_ab, stream>>>(X, A, w0h, w0l, b0, X4ws);

  reduce_over_i<<<dim3(128, 8), 128, 0, stream>>>(X4ws, X7m);

  size_t lds_cd = (size_t)(128 * LDP + 896 + 256) * 4;
  stage_cd<<<dim3(128, 8), 256, lds_cd, stream>>>(X, X4ws, Xdiag, X5m, X6m, X7m,
                                                  w10h, w10l, b10, W11, b11,
                                                  w2h, w2l, b2, out);
}
